// MixGraphExtractor_54065048322502
// MI455X (gfx1250) — matrix operand order repaired and checked by reading, not compiled, not run on hardware
//
#include <hip/hip_runtime.h>
#include <hip/hip_bf16.h>
#include <math.h>

#define NN 20000      // nodes
#define NE 320000     // edges
#define NG 64         // graphs
#define NR 5          // relations

typedef __attribute__((ext_vector_type(16))) _Float16     v16h;
typedef __attribute__((ext_vector_type(8)))  float        v8f;
typedef __attribute__((ext_vector_type(8)))  unsigned int v8u;

union Frag16 { v8u u; v16h h; };

// ---------------------------------------------------------------------------
// Generic WMMA GEMM: C[z] = act(A @ B[z] + bias), f32 in/out, f16 MACs.
// A: [N,K] row-major. B: [K,M] row-major (+ z*strideB). C: [N,M] (+ z*strideC)
// Block = 256 thr (8 waves), MUST be launched with exactly 256 threads.
// Block tile 32x64; wave tile 16x16.
// LDS: As row-major [row][k], Bs column-major [col][k]; K-pairs contiguous so
// fragments are straight b32 LDS loads.
// ---------------------------------------------------------------------------
__global__ __launch_bounds__(256)
void gemm_wmma_kernel(const float* __restrict__ A, const float* __restrict__ B,
                      const float* __restrict__ bias, float* __restrict__ C,
                      int N, int K, int M, long strideB, long strideC, int do_relu)
{
    __shared__ __align__(16) _Float16 As[32][42];   // 32 rows x 32 k (+pad, 21 dwords/row)
    __shared__ __align__(16) _Float16 Bs[64][42];   // 64 cols x 32 k (+pad)
    const int tid  = threadIdx.x;
    const int lane = tid & 31;
    const int wave = tid >> 5;
    const int rowBase = blockIdx.y * 32;
    const int colBase = blockIdx.x * 64;
    const float* Bp = B + (long)blockIdx.z * strideB;
    float*       Cp = C + (long)blockIdx.z * strideC;

    v8f acc = {0.f,0.f,0.f,0.f,0.f,0.f,0.f,0.f};

    const int wr   = (wave >> 2) << 4;   // wave row offset in block tile: 0/16
    const int wc   = (wave &  3) << 4;   // wave col offset: 0/16/32/48
    const int mrow = lane & 15;
    const int half = lane >> 4;          // lane half selects K sub-groups
    const int a0   = half * 4;           // A dword idx base: halves {0..7} vs {8..15}
    const int b0   = half * 8;           // unused; B fragment is indexed from a0 in the same k order as A

    for (int kk = 0; kk < K; kk += 32) {
        // ---- A tile: 32x32. Unconditional clamped loads, zeroing via x*mask
        // (0*Inf=NaN prevents the compiler from re-predicating the loads).
#pragma unroll
        for (int it = 0; it < 2; ++it) {
            int i  = tid + it * 256;                // 512 k-pairs total
            int r  = i >> 4;
            int c2 = (i & 15) * 2;
            int gr = rowBase + r;
            int gk = kk + c2;
            int grc = gr < N ? gr : N - 1;
            int k0c = gk     < K ? gk     : K - 1;
            int k1c = gk + 1 < K ? gk + 1 : K - 1;
            const float* ap = A + (long)grc * K;
            float m0 = ((gr < N) && (gk     < K)) ? 1.0f : 0.0f;
            float m1 = ((gr < N) && (gk + 1 < K)) ? 1.0f : 0.0f;
            float v0 = ap[k0c] * m0;
            float v1 = ap[k1c] * m1;
            unsigned h0 = __builtin_bit_cast(unsigned short, (_Float16)v0);
            unsigned h1 = __builtin_bit_cast(unsigned short, (_Float16)v1);
            ((unsigned*)&As[r][0])[c2 >> 1] = h0 | (h1 << 16);
        }
        // ---- B tile: 32x64 (global row-major, LDS column-major)
#pragma unroll
        for (int it = 0; it < 4; ++it) {
            int i  = tid + it * 256;                // 1024 k-pairs total
            int kp = i >> 6;                        // 0..15
            int c  = i & 63;                        // coalesced over columns
            int gk0 = kk + 2 * kp;
            int gk1 = gk0 + 1;
            int gc  = colBase + c;
            int gcc = gc  < M ? gc  : M - 1;
            int k0c = gk0 < K ? gk0 : K - 1;
            int k1c = gk1 < K ? gk1 : K - 1;
            float m0 = ((gk0 < K) && (gc < M)) ? 1.0f : 0.0f;
            float m1 = ((gk1 < K) && (gc < M)) ? 1.0f : 0.0f;
            float v0 = Bp[(long)k0c * M + gcc] * m0;
            float v1 = Bp[(long)k1c * M + gcc] * m1;
            unsigned h0 = __builtin_bit_cast(unsigned short, (_Float16)v0);
            unsigned h1 = __builtin_bit_cast(unsigned short, (_Float16)v1);
            ((unsigned*)&Bs[c][0])[kp] = h0 | (h1 << 16);
        }
        __syncthreads();

        // ---- fragments: pure b32 LDS loads (ISA 7.12.2 packed-pair layout)
        const unsigned* Arow = (const unsigned*)&As[wr + mrow][0];
        const unsigned* Brow = (const unsigned*)&Bs[wc + mrow][0];
        Frag16 a, b;
#pragma unroll
        for (int u = 0; u < 4; ++u) {
            a.u[u]     = Arow[a0 + u];       // K halves {ka0..ka0+7}
            a.u[4 + u] = Arow[8 + a0 + u];   // K halves {16+ka0..16+ka0+7}
        }
#pragma unroll
        for (int u = 0; u < 8; ++u) {
            b.u[u] = Brow[a0 + u + ((u >> 2) << 2)];   // K halves {ka0..ka0+7},{16+ka0..16+ka0+7}
        }
        acc = __builtin_amdgcn_wmma_f32_16x16x32_f16(
            false, a.h, false, b.h, (short)0, acc, false, false);
        __syncthreads();
    }

    // C/D layout: VGPR j -> row j (lanes 0-15) / row j+8 (lanes 16-31), col = lane&15
    const int gcol = colBase + wc + mrow;
    if (gcol < M) {
        const float bval = bias ? bias[gcol] : 0.0f;
        const int rofs = half * 8;
#pragma unroll
        for (int j = 0; j < 8; ++j) {
            int grow = rowBase + wr + rofs + j;
            if (grow < N) {
                float v = acc[j] + bval;
                if (do_relu) v = fmaxf(v, 0.0f);
                Cp[(long)grow * M + gcol] = v;
            }
        }
    }
}

// ---------------------------------------------------------------------------
// Elementwise / init kernels
// ---------------------------------------------------------------------------
__global__ void fill_f32_kernel(float* p, float v, long n) {
    long t = (long)blockIdx.x * blockDim.x + threadIdx.x;
    if (t < n) p[t] = v;
}
__global__ void fill_i32_kernel(int* p, int v, long n) {
    long t = (long)blockIdx.x * blockDim.x + threadIdx.x;
    if (t < n) p[t] = v;
}
__global__ void bias_act_kernel(float* x, const float* bias, long total, int F, int do_relu) {
    long t = (long)blockIdx.x * blockDim.x + threadIdx.x;
    if (t >= total) return;
    float v = x[t];
    if (bias) v += bias[t % F];
    if (do_relu) v = fmaxf(v, 0.0f);
    x[t] = v;
}

__device__ __forceinline__ void atomicMaxFloat(float* addr, float val) {
    if (val >= 0.0f) atomicMax((int*)addr, __float_as_int(val));
    else             atomicMin((unsigned int*)addr, __float_as_uint(val));
}

// ---------------------------------------------------------------------------
// RGCN pieces
// ---------------------------------------------------------------------------
__global__ void rgcn_count_kernel(const int* __restrict__ dst, const int* __restrict__ et,
                                  int* __restrict__ cnt, int E) {
    int e = blockIdx.x * blockDim.x + threadIdx.x;
    if (e < E) atomicAdd(&cnt[dst[e] * NR + et[e]], 1);
}
// grid.x = E, block covers F (rounded to 64)
__global__ void rgcn_scatter_kernel(const float* __restrict__ proj,
                                    const int* __restrict__ src, const int* __restrict__ dst,
                                    const int* __restrict__ et, const int* __restrict__ cnt,
                                    float* __restrict__ out, int F) {
    int e = blockIdx.x;
    int f = threadIdx.x;
    if (f >= F) return;
    int r = et[e], s = src[e], d = dst[e];
    int c = cnt[d * NR + r];
    float norm = 1.0f / (float)(c < 1 ? 1 : c);
    float v = proj[((long)r * NN + s) * F + f] * norm;
    atomicAdd(&out[(long)d * F + f], v);
}

// ---------------------------------------------------------------------------
// Edge encoder: e_emb = relu(edge_attr[E,8] @ We[8,16] + be)
// ---------------------------------------------------------------------------
__global__ void edge_enc_kernel(const float* __restrict__ ea, const float* __restrict__ W,
                                const float* __restrict__ b, float* __restrict__ out, int E) {
    int t = blockIdx.x * blockDim.x + threadIdx.x;
    if (t >= E * 16) return;
    int e = t >> 4, j = t & 15;
    float acc = b[j];
#pragma unroll
    for (int k = 0; k < 8; ++k) acc += ea[e * 8 + k] * W[k * 16 + j];
    out[t] = fmaxf(acc, 0.0f);
}

// ---------------------------------------------------------------------------
// GATv2 pieces (ee recomputed inline from e_emb, K=16)
// ---------------------------------------------------------------------------
__global__ void gat_logits_kernel(const float* __restrict__ xl, const float* __restrict__ xr,
                                  const float* __restrict__ eemb, const float* __restrict__ We,
                                  const float* __restrict__ att,
                                  const int* __restrict__ src, const int* __restrict__ dst,
                                  float* __restrict__ logits, float* __restrict__ mmax,
                                  int E, int H, int C) {
    int t = blockIdx.x * blockDim.x + threadIdx.x;
    if (t >= E * H) return;
    int e = t / H, h = t % H;
    int s = src[e], d = dst[e];
    const int HC = H * C;
    float e16[16];
#pragma unroll
    for (int k = 0; k < 16; ++k) e16[k] = eemb[(long)e * 16 + k];
    const float* xls  = xl + (long)s * HC + h * C;
    const float* xrd  = xr + (long)d * HC + h * C;
    const float* atth = att + h * C;
    const float* Weh  = We + h * C;
    float acc = 0.0f;
    for (int c = 0; c < C; ++c) {
        float ee = 0.0f;
#pragma unroll
        for (int k = 0; k < 16; ++k) ee += e16[k] * Weh[k * HC + c];
        float z = xls[c] + xrd[c] + ee;
        z = (z > 0.0f) ? z : 0.2f * z;          // leaky_relu(0.2)
        acc += atth[c] * z;
    }
    logits[t] = acc;
    atomicMaxFloat(&mmax[d * H + h], acc);
}

__global__ void gat_exp_kernel(const float* __restrict__ logits, const float* __restrict__ mmax,
                               const int* __restrict__ dst, float* __restrict__ aexp,
                               float* __restrict__ denom, int E, int H) {
    int t = blockIdx.x * blockDim.x + threadIdx.x;
    if (t >= E * H) return;
    int e = t / H, h = t % H;
    int d = dst[e];
    float a = expf(logits[t] - mmax[d * H + h]);
    aexp[t] = a;
    atomicAdd(&denom[d * H + h], a);
}

// grid.x = E, block covers H*C (rounded to 64)
__global__ void gat_scatter_kernel(const float* __restrict__ xl, const float* __restrict__ aexp,
                                   const float* __restrict__ denom,
                                   const int* __restrict__ src, const int* __restrict__ dst,
                                   float* __restrict__ out, int H, int C) {
    int e = blockIdx.x;
    int f = threadIdx.x;
    const int HC = H * C;
    if (f >= HC) return;
    int s = src[e], d = dst[e];
    int h = f / C;
    float w = aexp[e * H + h] / denom[d * H + h];
    atomicAdd(&out[(long)d * HC + f], xl[(long)s * HC + f] * w);
}

// ---------------------------------------------------------------------------
// Pooling: p[g] = [max | sum/cnt], layout [G][2F]
// ---------------------------------------------------------------------------
__global__ void init_pool_kernel(float* p, int G, int F) {
    int t = blockIdx.x * blockDim.x + threadIdx.x;
    if (t >= G * 2 * F) return;
    int col = t % (2 * F);
    p[t] = (col < F) ? __int_as_float(0xff800000) : 0.0f;   // -inf | 0
}
__global__ void node_count_kernel(const int* __restrict__ batch, int* __restrict__ gcnt, int N) {
    int t = blockIdx.x * blockDim.x + threadIdx.x;
    if (t < N) atomicAdd(&gcnt[batch[t]], 1);
}
// grid.x = N, block covers F (rounded to 64)
__global__ void pool_scatter_kernel(const float* __restrict__ h, const int* __restrict__ batch,
                                    float* __restrict__ p, int F) {
    int n = blockIdx.x;
    int f = threadIdx.x;
    if (f >= F) return;
    int g = batch[n];
    float v = h[(long)n * F + f];
    atomicMaxFloat(&p[(long)g * 2 * F + f], v);
    atomicAdd(&p[(long)g * 2 * F + F + f], v);
}
__global__ void pool_div_kernel(float* p, const int* __restrict__ gcnt, int G, int F) {
    int t = blockIdx.x * blockDim.x + threadIdx.x;
    if (t >= G * F) return;
    int g = t / F, f = t % F;
    p[(long)g * 2 * F + F + f] /= (float)gcnt[g];
}

__global__ void concat_kernel(const float* __restrict__ p1, const float* __restrict__ p2,
                              float* __restrict__ z, int G) {
    int t = blockIdx.x * blockDim.x + threadIdx.x;
    if (t >= G * 1872) return;
    int g = t / 1872, j = t % 1872;
    z[t] = (j < 624) ? p1[g * 624 + j] : p2[g * 1248 + (j - 624)];
}

// ---------------------------------------------------------------------------
// LayerNorm over M=1024, one block per row (relu already applied by GEMM)
// ---------------------------------------------------------------------------
__global__ __launch_bounds__(256)
void layernorm_kernel(const float* __restrict__ z, const float* __restrict__ g,
                      const float* __restrict__ b, float* __restrict__ out, int M) {
    __shared__ float red[256];
    int row = blockIdx.x, tid = threadIdx.x;
    const float* zr = z + (long)row * M;
    float s = 0.0f;
    for (int i = tid; i < M; i += 256) s += zr[i];
    red[tid] = s; __syncthreads();
    for (int w = 128; w > 0; w >>= 1) { if (tid < w) red[tid] += red[tid + w]; __syncthreads(); }
    float mu = red[0] / (float)M;
    __syncthreads();
    float s2 = 0.0f;
    for (int i = tid; i < M; i += 256) { float d = zr[i] - mu; s2 += d * d; }
    red[tid] = s2; __syncthreads();
    for (int w = 128; w > 0; w >>= 1) { if (tid < w) red[tid] += red[tid + w]; __syncthreads(); }
    float inv = rsqrtf(red[0] / (float)M + 1e-5f);
    for (int i = tid; i < M; i += 256)
        out[(long)row * M + i] = (zr[i] - mu) * inv * g[i] + b[i];
}

// ---------------------------------------------------------------------------
// Host side
// ---------------------------------------------------------------------------
static inline int rup(int x, int a) { return ((x + a - 1) / a) * a; }

static void launch_gemm(const float* A, const float* B, const float* bias, float* C,
                        int N, int K, int M, int batch, long sB, long sC, int relu,
                        hipStream_t s) {
    dim3 grid((M + 63) / 64, (N + 31) / 32, batch);
    gemm_wmma_kernel<<<grid, 256, 0, s>>>(A, B, bias, C, N, K, M, sB, sC, relu);
}
static void fillf(float* p, float v, long n, hipStream_t s) {
    fill_f32_kernel<<<(int)((n + 255) / 256), 256, 0, s>>>(p, v, n);
}
static void filli(int* p, int v, long n, hipStream_t s) {
    fill_i32_kernel<<<(int)((n + 255) / 256), 256, 0, s>>>(p, v, n);
}

extern "C" void kernel_launch(void* const* d_in, const int* in_sizes, int n_in,
                              void* d_out, int out_size, void* d_ws, size_t ws_size,
                              hipStream_t stream) {
    (void)in_sizes; (void)n_in; (void)out_size; (void)ws_size;
    const float* x         = (const float*)d_in[0];
    const int*   eidx      = (const int*)  d_in[1];
    const float* edge_attr = (const float*)d_in[2];
    const int*   etype     = (const int*)  d_in[3];
    const int*   batch     = (const int*)  d_in[4];
    const float* We_enc    = (const float*)d_in[5];
    const float* be_enc    = (const float*)d_in[6];
    const float* Wrel1 = (const float*)d_in[7];  const float* Wroot1 = (const float*)d_in[8];  const float* b1 = (const float*)d_in[9];
    const float* Wrel2 = (const float*)d_in[10]; const float* Wroot2 = (const float*)d_in[11]; const float* b2 = (const float*)d_in[12];
    const float* Wrel3 = (const float*)d_in[13]; const float* Wroot3 = (const float*)d_in[14]; const float* b3 = (const float*)d_in[15];
    const float* Wl1 = (const float*)d_in[16]; const float* Wr1 = (const float*)d_in[17];
    const float* We1 = (const float*)d_in[18]; const float* att1 = (const float*)d_in[19]; const float* bg1 = (const float*)d_in[20];
    const float* Wl2 = (const float*)d_in[21]; const float* Wr2 = (const float*)d_in[22];
    const float* We2 = (const float*)d_in[23]; const float* att2 = (const float*)d_in[24]; const float* bg2 = (const float*)d_in[25];
    const float* Wfc = (const float*)d_in[26]; const float* bfc = (const float*)d_in[27];
    const float* ln_g = (const float*)d_in[28]; const float* ln_b = (const float*)d_in[29];

    const int* src = eidx;            // edge_index[0]
    const int* dst = eidx + NE;       // edge_index[1]

    // ---- workspace layout (256B aligned) ----
    char* ws = (char*)d_ws;
    size_t off = 0;
    auto alloc = [&](size_t bytes) { size_t o = off; off += (bytes + 255) & ~(size_t)255; return o; };
    int*   cnt    = (int*)  (ws + alloc((size_t)NN * NR * 4));
    float* e_emb  = (float*)(ws + alloc((size_t)NE * 16 * 4));
    float* proj   = (float*)(ws + alloc((size_t)NR * NN * 312 * 4));   // also reused as xl/xr
    float* h1a    = (float*)(ws + alloc((size_t)NN * 78  * 4));
    float* h1b    = (float*)(ws + alloc((size_t)NN * 156 * 4));
    float* h1c    = (float*)(ws + alloc((size_t)NN * 312 * 4));
    float* logits = (float*)(ws + alloc((size_t)NE * 4 * 4));
    float* aexp   = (float*)(ws + alloc((size_t)NE * 4 * 4));
    float* mmax   = (float*)(ws + alloc((size_t)NN * 4 * 4));
    float* denom  = (float*)(ws + alloc((size_t)NN * 4 * 4));
    float* h2a    = (float*)(ws + alloc((size_t)NN * 156 * 4));
    float* h2b    = (float*)(ws + alloc((size_t)NN * 624 * 4));
    float* p1     = (float*)(ws + alloc((size_t)NG * 624  * 4));
    float* p2     = (float*)(ws + alloc((size_t)NG * 1248 * 4));
    int*   gcnt   = (int*)  (ws + alloc((size_t)NG * 4));
    float* zfeat  = (float*)(ws + alloc((size_t)NG * 1872 * 4));
    float* zlin   = (float*)(ws + alloc((size_t)NG * 1024 * 4));
    float* xl = proj;                         // GAT reuse of proj region
    float* xr = proj + (size_t)NN * 624;

    // ---- per-(dst,rel) edge counts (shared by all RGCN layers) ----
    filli(cnt, 0, (long)NN * NR, stream);
    rgcn_count_kernel<<<(NE + 255) / 256, 256, 0, stream>>>(dst, etype, cnt, NE);

    // ---- RGCN layer 1: 78 -> 78 ----
    launch_gemm(x, Wrel1, nullptr, proj, NN, 78, 78, NR, 78L * 78, (long)NN * 78, 0, stream);
    launch_gemm(x, Wroot1, b1, h1a, NN, 78, 78, 1, 0, 0, 0, stream);
    rgcn_scatter_kernel<<<NE, rup(78, 64), 0, stream>>>(proj, src, dst, etype, cnt, h1a, 78);
    bias_act_kernel<<<(int)(((long)NN * 78 + 255) / 256), 256, 0, stream>>>(h1a, nullptr, (long)NN * 78, 78, 1);

    // ---- RGCN layer 2: 78 -> 156 ----
    launch_gemm(h1a, Wrel2, nullptr, proj, NN, 78, 156, NR, 78L * 156, (long)NN * 156, 0, stream);
    launch_gemm(h1a, Wroot2, b2, h1b, NN, 78, 156, 1, 0, 0, 0, stream);
    rgcn_scatter_kernel<<<NE, rup(156, 64), 0, stream>>>(proj, src, dst, etype, cnt, h1b, 156);
    bias_act_kernel<<<(int)(((long)NN * 156 + 255) / 256), 256, 0, stream>>>(h1b, nullptr, (long)NN * 156, 156, 1);

    // ---- RGCN layer 3: 156 -> 312 ----
    launch_gemm(h1b, Wrel3, nullptr, proj, NN, 156, 312, NR, 156L * 312, (long)NN * 312, 0, stream);
    launch_gemm(h1b, Wroot3, b3, h1c, NN, 156, 312, 1, 0, 0, 0, stream);
    rgcn_scatter_kernel<<<NE, rup(312, 64), 0, stream>>>(proj, src, dst, etype, cnt, h1c, 312);
    bias_act_kernel<<<(int)(((long)NN * 312 + 255) / 256), 256, 0, stream>>>(h1c, nullptr, (long)NN * 312, 312, 1);

    // ---- pool RGCN branch -> p1 [G, 624] = [max | mean] ----
    filli(gcnt, 0, NG, stream);
    node_count_kernel<<<(NN + 255) / 256, 256, 0, stream>>>(batch, gcnt, NN);
    init_pool_kernel<<<(NG * 624 + 255) / 256, 256, 0, stream>>>(p1, NG, 312);
    pool_scatter_kernel<<<NN, rup(312, 64), 0, stream>>>(h1c, batch, p1, 312);
    pool_div_kernel<<<(NG * 312 + 255) / 256, 256, 0, stream>>>(p1, gcnt, NG, 312);

    // ---- edge encoder ----
    edge_enc_kernel<<<(NE * 16 + 255) / 256, 256, 0, stream>>>(edge_attr, We_enc, be_enc, e_emb, NE);

    // ---- GATv2 layer 1: H=2, C=78 ----
    launch_gemm(x, Wl1, nullptr, xl, NN, 78, 156, 1, 0, 0, 0, stream);
    launch_gemm(x, Wr1, nullptr, xr, NN, 78, 156, 1, 0, 0, 0, stream);
    fillf(mmax, -INFINITY, (long)NN * 2, stream);
    fillf(denom, 0.0f, (long)NN * 2, stream);
    fillf(h2a, 0.0f, (long)NN * 156, stream);
    gat_logits_kernel<<<(NE * 2 + 255) / 256, 256, 0, stream>>>(xl, xr, e_emb, We1, att1, src, dst, logits, mmax, NE, 2, 78);
    gat_exp_kernel<<<(NE * 2 + 255) / 256, 256, 0, stream>>>(logits, mmax, dst, aexp, denom, NE, 2);
    gat_scatter_kernel<<<NE, rup(156, 64), 0, stream>>>(xl, aexp, denom, src, dst, h2a, 2, 78);
    bias_act_kernel<<<(int)(((long)NN * 156 + 255) / 256), 256, 0, stream>>>(h2a, bg1, (long)NN * 156, 156, 1);

    // ---- GATv2 layer 2: H=4, C=156 ----
    launch_gemm(h2a, Wl2, nullptr, xl, NN, 156, 624, 1, 0, 0, 0, stream);
    launch_gemm(h2a, Wr2, nullptr, xr, NN, 156, 624, 1, 0, 0, 0, stream);
    fillf(mmax, -INFINITY, (long)NN * 4, stream);
    fillf(denom, 0.0f, (long)NN * 4, stream);
    fillf(h2b, 0.0f, (long)NN * 624, stream);
    gat_logits_kernel<<<(NE * 4 + 255) / 256, 256, 0, stream>>>(xl, xr, e_emb, We2, att2, src, dst, logits, mmax, NE, 4, 156);
    gat_exp_kernel<<<(NE * 4 + 255) / 256, 256, 0, stream>>>(logits, mmax, dst, aexp, denom, NE, 4);
    gat_scatter_kernel<<<NE, rup(624, 64), 0, stream>>>(xl, aexp, denom, src, dst, h2b, 4, 156);
    bias_act_kernel<<<(int)(((long)NN * 624 + 255) / 256), 256, 0, stream>>>(h2b, bg2, (long)NN * 624, 624, 1);

    // ---- pool GAT branch -> p2 [G, 1248] ----
    init_pool_kernel<<<(NG * 1248 + 255) / 256, 256, 0, stream>>>(p2, NG, 624);
    pool_scatter_kernel<<<NN, rup(624, 64), 0, stream>>>(h2b, batch, p2, 624);
    pool_div_kernel<<<(NG * 624 + 255) / 256, 256, 0, stream>>>(p2, gcnt, NG, 624);

    // ---- head: concat -> Linear(1872,1024)+ReLU (WMMA) -> LayerNorm ----
    concat_kernel<<<(NG * 1872 + 255) / 256, 256, 0, stream>>>(p1, p2, zfeat, NG);
    launch_gemm(zfeat, Wfc, bfc, zlin, NG, 1872, 1024, 1, 0, 0, 1, stream);
    layernorm_kernel<<<NG, 256, 0, stream>>>(zlin, ln_g, ln_b, (float*)d_out, 1024);
}
